// SwinTransformerBlock_58780922413542
// MI455X (gfx1250) — matrix operand order repaired and checked by reading, not compiled, not run on hardware
//
#include <hip/hip_runtime.h>
#include <hip/hip_bf16.h>

typedef __attribute__((ext_vector_type(16))) _Float16 v16h;
typedef __attribute__((ext_vector_type(8)))  float    v8f;
typedef __attribute__((ext_vector_type(8)))  _Float16 h8;

constexpr int kBL  = 131072;  // B * H * W tokens
constexpr int kDIM = 256;

__device__ inline float gelu_exact(float x) {
    return 0.5f * x * (1.0f + erff(x * 0.70710678118654752f));
}

// ---- LDS helpers -----------------------------------------------------------
typedef __attribute__((address_space(3))) const char lds_cchar;
__device__ inline uint32_t to_lds_off(const void* p) {
    return (uint32_t)(uintptr_t)(lds_cchar*)p;
}
// Async DMA: 16B per lane, global -> LDS, tracked by ASYNCcnt (CDNA5 §10.6/15.18).
__device__ inline void async_copy_b128(uint32_t lds_off, const _Float16* g) {
    asm volatile("global_load_async_to_lds_b128 %0, %1, off"
                 :: "v"(lds_off), "v"(g) : "memory");
}
#define WAIT_ASYNCCNT(n) asm volatile("s_wait_asynccnt %0" :: "i"(n) : "memory")

#if __has_builtin(__builtin_amdgcn_s_wait_tensorcnt)
#define WAIT_TENSORCNT(n) __builtin_amdgcn_s_wait_tensorcnt((unsigned short)(n))
#else
#define WAIT_TENSORCNT(n) asm volatile("s_wait_tensorcnt %0" :: "i"(n) : "memory")
#endif

// ---- Tensor Data Mover: one-instruction DMA of a 2D tile into LDS ----------
#if __has_builtin(__builtin_amdgcn_tensor_load_to_lds)
#define HAS_TDM 1
typedef unsigned int u32x4 __attribute__((ext_vector_type(4)));
typedef int          i32x8 __attribute__((ext_vector_type(8)));
typedef int          i32x4 __attribute__((ext_vector_type(4)));
// Tile: 32 halves x 128 rows, row stride = strideElems halves; DMA inserts 16B LDS pad
// after every 64B row (pad_interval=16 dwords, pad_amount=4 dwords) -> 80B LDS rows.
__device__ inline void tdm_load_tile_128x32(uint32_t lds_off, const _Float16* g,
                                            uint32_t strideElems) {
    const uint64_t ga = (uint64_t)(uintptr_t)g;
    u32x4 g0;
    g0[0] = 1u;                                  // D# group0: count=1 (valid user desc)
    g0[1] = lds_off;                             // lds_addr (bytes)
    g0[2] = (uint32_t)ga;                        // global_addr[31:0] (tile start)
    g0[3] = (uint32_t)(ga >> 32) | (2u << 30);   // global_addr[56:32] | type=2 ("image")
    i32x8 g1;
    g1[0] = (int)((1u << 16) | (1u << 20) | (3u << 22) | (3u << 25));
        // data_size=2B | pad_enable | pad_interval: 16 dwords | pad_amount: 4 dwords
    g1[1] = (int)(32u << 16);                    // tensor_dim0 = 32 (halves)
    g1[2] = (int)(128u << 16);                   // tensor_dim1 = 128
    g1[3] = (int)(32u << 16);                    // tile_dim0 = 32
    g1[4] = 128;                                 // tile_dim1 = 128, tile_dim2 = 0
    g1[5] = (int)strideElems;                    // tensor_dim0_stride[31:0]
    g1[6] = 0;                                   // stride[47:32]=0, dim1_stride lo=0
    g1[7] = 0;
    const i32x4 zz4 = {0, 0, 0, 0};
    const i32x8 zz8 = {0, 0, 0, 0, 0, 0, 0, 0};
    // 6-arg form on this toolchain: (g0, g1, g2, g3, extra, cpol)
    __builtin_amdgcn_tensor_load_to_lds(g0, g1, zz4, zz4, zz8, 0);
}
#else
#define HAS_TDM 0
#endif

// Build a 16-half fragment from two 16B chunks (two ds_load_b128 / global_load_b128).
__device__ inline v16h ld_frag2(const _Float16* p0, const _Float16* p1) {
    union { v16h v; h8 h[2]; } u;
    u.h[0] = *(const h8*)p0;
    u.h[1] = *(const h8*)p1;
    return u.v;
}

// ---------------- f32 (K x N) -> f16 transposed (N x K) ----------------
__global__ void cvt_f16_t_kernel(const float* __restrict__ src, _Float16* __restrict__ dst,
                                 int K, int N) {
    int i = blockIdx.x * blockDim.x + threadIdx.x;
    if (i < K * N) {
        int k = i / N, n = i - k * N;
        dst[(size_t)n * K + k] = (_Float16)src[i];
    }
}

// ---------------- LayerNorm over 256, one wave per row, f16 out ----------------
__global__ __launch_bounds__(128)
void ln_f16_kernel(const float* __restrict__ x, const float* __restrict__ g,
                   const float* __restrict__ b, _Float16* __restrict__ out) {
    const int wave = threadIdx.x >> 5;
    const int lane = threadIdx.x & 31;
    const size_t row = (size_t)blockIdx.x * 4 + wave;
    const float* p = x + row * kDIM + lane * 8;
    float v[8];
    float s = 0.f, ss = 0.f;
#pragma unroll
    for (int i = 0; i < 8; ++i) { v[i] = p[i]; s += v[i]; ss += v[i] * v[i]; }
#pragma unroll
    for (int off = 16; off; off >>= 1) {
        s  += __shfl_xor(s,  off, 32);
        ss += __shfl_xor(ss, off, 32);
    }
    const float mu  = s * (1.0f / kDIM);
    const float var = ss * (1.0f / kDIM) - mu * mu;
    const float inv = rsqrtf(var + 1e-5f);
    _Float16* o = out + row * kDIM + lane * 8;
#pragma unroll
    for (int i = 0; i < 8; ++i)
        o[i] = (_Float16)((v[i] - mu) * inv * g[lane * 8 + i] + b[lane * 8 + i]);
}

// ---------------- WMMA GEMM: C[M,N] = A[M,K](f16) @ Wt[N,K](f16)^T + bias ----------------
// 128x128 block tile, K-tile 32, double-buffered.
// A tile: per-lane GLOBAL_LOAD_ASYNC_TO_LDS_B128 (ASYNCcnt).
// B tile: single TENSOR_LOAD_TO_LDS issued by wave 0 (TENSORcnt), DMA-inserted LDS pad.
template <int ACT_GELU, int OUT_F16, int RESID>
__global__ __launch_bounds__(256)
void gemm_wmma_kernel(const _Float16* __restrict__ A, const _Float16* __restrict__ Wt,
                      const float* __restrict__ bias, const float* __restrict__ resid,
                      void* __restrict__ outp, int M, int N, int K) {
    constexpr int LD = 40;  // halves: 80B rows, 16B aligned
    __shared__ _Float16 As[2][128 * LD];
    __shared__ _Float16 Bs[2][128 * LD];

    const int tid  = threadIdx.x;
    const int lane = tid & 31;
    const int wid  = tid >> 5;
    const int wm   = wid & 1;
    const int wn   = wid >> 1;
    const int l16  = lane & 15;
    const int kh   = lane >> 4;
    const int m0   = blockIdx.x * 128;
    const int n0   = blockIdx.y * 128;

    const int r  = tid >> 1;        // 0..127: A-tile row handled by this thread
    const int c0 = (tid & 1) << 4;  // 0 or 16: K-half chunk base

    auto issue_tile = [&](int kt, int buf) {
        const int k0 = kt << 5;
        async_copy_b128(to_lds_off(&As[buf][r * LD + c0]),
                        A + (size_t)(m0 + r) * K + k0 + c0);
        async_copy_b128(to_lds_off(&As[buf][r * LD + c0 + 8]),
                        A + (size_t)(m0 + r) * K + k0 + c0 + 8);
#if HAS_TDM
        if (wid == 0)
            tdm_load_tile_128x32(to_lds_off(&Bs[buf][0]),
                                 Wt + (size_t)n0 * K + k0, (uint32_t)K);
#else
        async_copy_b128(to_lds_off(&Bs[buf][r * LD + c0]),
                        Wt + (size_t)(n0 + r) * K + k0 + c0);
        async_copy_b128(to_lds_off(&Bs[buf][r * LD + c0 + 8]),
                        Wt + (size_t)(n0 + r) * K + k0 + c0 + 8);
#endif
    };
#if HAS_TDM
    constexpr int kAsyncInFlight = 2;
#else
    constexpr int kAsyncInFlight = 4;
#endif

    const v8f zero8 = {0.f, 0.f, 0.f, 0.f, 0.f, 0.f, 0.f, 0.f};
    v8f acc[4][2];
#pragma unroll
    for (int mi = 0; mi < 4; ++mi)
#pragma unroll
        for (int nj = 0; nj < 2; ++nj) acc[mi][nj] = zero8;

    const int nk = K >> 5;
    issue_tile(0, 0);
    for (int kt = 0; kt < nk; ++kt) {
        if (kt + 1 < nk) {
            issue_tile(kt + 1, (kt + 1) & 1);
            WAIT_ASYNCCNT(kAsyncInFlight);   // in-order: current tile's async copies landed
            WAIT_TENSORCNT(1);               // current tile's TDM landed (0 for non-issuers)
        } else {
            WAIT_ASYNCCNT(0);
            WAIT_TENSORCNT(0);
        }
        __syncthreads();   // tile resident in LDS for all waves

        const _Float16* as = &As[kt & 1][0];
        const _Float16* bs = &Bs[kt & 1][0];
        v16h a[4], b[2];
#pragma unroll
        for (int mi = 0; mi < 4; ++mi) {
            const _Float16* p = as + (wm * 64 + mi * 16 + l16) * LD + kh * 8;
            a[mi] = ld_frag2(p, p + 16);
        }
#pragma unroll
        for (int nj = 0; nj < 2; ++nj) {
            const _Float16* p = bs + (wn * 32 + nj * 16 + l16) * LD + kh * 8;
            b[nj] = ld_frag2(p, p + 16);
        }
#pragma unroll
        for (int mi = 0; mi < 4; ++mi)
#pragma unroll
            for (int nj = 0; nj < 2; ++nj)
                acc[mi][nj] = __builtin_amdgcn_wmma_f32_16x16x32_f16(
                    false, a[mi], false, b[nj], (short)0, acc[mi][nj], false, false);
        __syncthreads();   // done reading this buffer before it is overwritten
    }

#pragma unroll
    for (int mi = 0; mi < 4; ++mi)
#pragma unroll
        for (int nj = 0; nj < 2; ++nj)
#pragma unroll
            for (int i = 0; i < 8; ++i) {
                const size_t row = (size_t)m0 + wm * 64 + mi * 16 + i + 8 * kh;
                const int col = n0 + wn * 32 + nj * 16 + l16;
                float v = acc[mi][nj][i] + bias[col];
                if (ACT_GELU) v = gelu_exact(v);
                if (RESID) v += resid[row * N + col];
                if (OUT_F16) ((_Float16*)outp)[row * N + col] = (_Float16)v;
                else         ((float*)outp)[row * N + col]    = v;
            }
}

// ---------------- Window attention: one wave per (window, head) ----------------
__global__ __launch_bounds__(64)
void attn_kernel(const _Float16* __restrict__ qkv, const float* __restrict__ rel_bias,
                 _Float16* __restrict__ out) {
    __shared__ float    S[2][64][68];     // 272B f32 rows -> f16 P rows stay 16B aligned
    __shared__ _Float16 Vt[2][32 * 80];   // V transposed [d][tok], 160B rows
    const int tid  = threadIdx.x;
    const int lane = tid & 31;
    const int sl   = tid >> 5;
    const int id   = blockIdx.x * 2 + sl;
    const int w    = id >> 3;
    const int h    = id & 7;
    const int l16  = lane & 15;
    const int kh   = lane >> 4;
    const int bW = w >> 8, rem = w & 255, wi = rem >> 4, wj = rem & 15;
    auto tok = [&](int n) -> size_t {
        return (size_t)bW * 16384 + (size_t)(wi * 8 + (n >> 3)) * 128 + wj * 8 + (n & 7);
    };

    // stage V transposed into LDS: each lane handles 2 tokens (contig 32-half rows in)
#pragma unroll
    for (int t2 = 0; t2 < 2; ++t2) {
        const int t = lane * 2 + t2;
        const _Float16* pv = qkv + tok(t) * 768 + 512 + h * 32;
        h8 vv[4];
#pragma unroll
        for (int q = 0; q < 4; ++q) vv[q] = *(const h8*)(pv + q * 8);
#pragma unroll
        for (int d = 0; d < 32; ++d) Vt[sl][d * 80 + t] = vv[d >> 3][d & 7];
    }

    // ---- S = (q * scale) @ k^T  (64x64, K=32) ----
    v16h aq[4], bk[4];
#pragma unroll
    for (int mi = 0; mi < 4; ++mi) {
        const _Float16* p = qkv + tok(mi * 16 + l16) * 768 + h * 32 + kh * 8;
        aq[mi] = ld_frag2(p, p + 16);
    }
#pragma unroll
    for (int nj = 0; nj < 4; ++nj) {
        const _Float16* p = qkv + tok(nj * 16 + l16) * 768 + 256 + h * 32 + kh * 8;
        bk[nj] = ld_frag2(p, p + 16);
    }
    const v8f zero8 = {0.f, 0.f, 0.f, 0.f, 0.f, 0.f, 0.f, 0.f};
    const float scale = 0.17677669529663687f;  // 1/sqrt(32)
#pragma unroll
    for (int mi = 0; mi < 4; ++mi)
#pragma unroll
        for (int nj = 0; nj < 4; ++nj) {
            v8f c = __builtin_amdgcn_wmma_f32_16x16x32_f16(
                false, aq[mi], false, bk[nj], (short)0, zero8, false, false);
#pragma unroll
            for (int i = 0; i < 8; ++i) {
                const int qn = mi * 16 + i + 8 * kh;
                const int km = nj * 16 + l16;
                const int ridx = (((qn >> 3) - (km >> 3) + 7) * 15) + ((qn & 7) - (km & 7) + 7);
                S[sl][qn][km] = c[i] * scale + rel_bias[ridx * 8 + h];
            }
        }
    __syncthreads();

    // ---- softmax rows, P (f16) written in place over each row start ----
    for (int rr = tid; rr < 128; rr += 64) {
        const int s2 = rr >> 6, r = rr & 63;
        float* srow = &S[s2][r][0];
        float mx = srow[0];
#pragma unroll 8
        for (int j = 1; j < 64; ++j) mx = fmaxf(mx, srow[j]);
        float sum = 0.f;
#pragma unroll 8
        for (int j = 0; j < 64; ++j) sum += __expf(srow[j] - mx);
        const float inv = 1.0f / sum;
        _Float16* prow = (_Float16*)srow;
        for (int j = 0; j < 64; ++j)           // write byte 2j < read byte 4j: safe in place
            prow[j] = (_Float16)(__expf(srow[j] - mx) * inv);
    }
    __syncthreads();

    // ---- O = P @ V  (64x32, K=64 -> two K-steps) ----
    v8f o[4][2];
#pragma unroll
    for (int mi = 0; mi < 4; ++mi)
#pragma unroll
        for (int nj = 0; nj < 2; ++nj) o[mi][nj] = zero8;
#pragma unroll
    for (int kt = 0; kt < 2; ++kt) {
        v16h pa[4], vb[2];
#pragma unroll
        for (int mi = 0; mi < 4; ++mi) {
            const _Float16* p = (const _Float16*)&S[sl][mi * 16 + l16][0] + kt * 32 + kh * 8;
            pa[mi] = ld_frag2(p, p + 16);
        }
#pragma unroll
        for (int nj = 0; nj < 2; ++nj) {
            const _Float16* p = &Vt[sl][(nj * 16 + l16) * 80 + kt * 32 + kh * 8];
            vb[nj] = ld_frag2(p, p + 16);
        }
#pragma unroll
        for (int mi = 0; mi < 4; ++mi)
#pragma unroll
            for (int nj = 0; nj < 2; ++nj)
                o[mi][nj] = __builtin_amdgcn_wmma_f32_16x16x32_f16(
                    false, pa[mi], false, vb[nj], (short)0, o[mi][nj], false, false);
    }
#pragma unroll
    for (int mi = 0; mi < 4; ++mi)
#pragma unroll
        for (int nj = 0; nj < 2; ++nj)
#pragma unroll
            for (int i = 0; i < 8; ++i) {
                const int qn = mi * 16 + i + 8 * kh;
                const int d  = nj * 16 + l16;
                out[tok(qn) * 256 + h * 32 + d] = (_Float16)o[mi][nj][i];
            }
}

// ---------------- 3x3 depthwise conv (NHWC, C=1024) + GELU, f16 in/out ----------------
__global__ __launch_bounds__(256)
void dwconv_gelu_kernel(const _Float16* __restrict__ y, const float* __restrict__ wgt,
                        const float* __restrict__ bias, _Float16* __restrict__ o) {
    const size_t idx = (size_t)blockIdx.x * 256 + threadIdx.x;
    const int c = (int)(idx & 1023);
    const size_t pix = idx >> 10;
    const int px = (int)(pix & 127);
    const int py = (int)((pix >> 7) & 127);
    const int pb = (int)(pix >> 14);
    float acc = bias[c];
#pragma unroll
    for (int ky = 0; ky < 3; ++ky) {
        const int iy = py + ky - 1;
        if (iy < 0 || iy > 127) continue;
#pragma unroll
        for (int kx = 0; kx < 3; ++kx) {
            const int ix = px + kx - 1;
            if (ix < 0 || ix > 127) continue;
            acc += (float)y[(((size_t)pb * 128 + iy) * 128 + ix) * 1024 + c] * wgt[c * 9 + ky * 3 + kx];
        }
    }
    o[idx] = (_Float16)gelu_exact(acc);
}

// ---------------- launch ----------------
extern "C" void kernel_launch(void* const* d_in, const int* in_sizes, int n_in,
                              void* d_out, int out_size, void* d_ws, size_t ws_size,
                              hipStream_t stream) {
    const float* x      = (const float*)d_in[0];
    const float* n1g    = (const float*)d_in[3];
    const float* n1b    = (const float*)d_in[4];
    const float* qkv_w  = (const float*)d_in[5];
    const float* qkv_b  = (const float*)d_in[6];
    const float* relb   = (const float*)d_in[7];
    const float* proj_w = (const float*)d_in[8];
    const float* proj_b = (const float*)d_in[9];
    const float* n2g    = (const float*)d_in[10];
    const float* n2b    = (const float*)d_in[11];
    const float* l1w    = (const float*)d_in[12];
    const float* l1b    = (const float*)d_in[13];
    const float* dww    = (const float*)d_in[14];
    const float* dwb    = (const float*)d_in[15];
    const float* l2w    = (const float*)d_in[16];
    const float* l2b    = (const float*)d_in[17];
    float* out = (float*)d_out;

    char* ws = (char*)d_ws;
    size_t off = 0;
    auto alloc = [&](size_t bytes) -> void* {
        void* p = ws + off;
        off = (off + bytes + 255) & ~(size_t)255;
        return p;
    };
    // transposed f16 weights: (N x K)
    _Float16* wqkv_t  = (_Float16*)alloc((size_t)768 * 256 * 2);
    _Float16* wproj_t = (_Float16*)alloc((size_t)256 * 256 * 2);
    _Float16* wl1_t   = (_Float16*)alloc((size_t)1024 * 256 * 2);
    _Float16* wl2_t   = (_Float16*)alloc((size_t)256 * 1024 * 2);
    _Float16* xn_h    = (_Float16*)alloc((size_t)kBL * 256 * 2);   // LN1 out, reused for LN2 out
    _Float16* qkv_h   = (_Float16*)alloc((size_t)kBL * 768 * 2);   // } contiguous pair reused
    _Float16* attn_h  = (_Float16*)alloc((size_t)kBL * 256 * 2);   // }   as y1 (kBL x 1024)
    float*    proj_f  = (float*)   alloc((size_t)kBL * 256 * 4);
    _Float16* y2_h    = (_Float16*)alloc((size_t)kBL * 1024 * 2);
    _Float16* y1_h    = qkv_h;  // 768+256 = 1024 halves/row overlay, safe after attention

    // 1. weights -> f16, transposed to N x K
    cvt_f16_t_kernel<<<(256 * 768 + 255) / 256, 256, 0, stream>>>(qkv_w, wqkv_t, 256, 768);
    cvt_f16_t_kernel<<<(256 * 256 + 255) / 256, 256, 0, stream>>>(proj_w, wproj_t, 256, 256);
    cvt_f16_t_kernel<<<(256 * 1024 + 255) / 256, 256, 0, stream>>>(l1w, wl1_t, 256, 1024);
    cvt_f16_t_kernel<<<(1024 * 256 + 255) / 256, 256, 0, stream>>>(l2w, wl2_t, 1024, 256);

    // 2. LN1
    ln_f16_kernel<<<kBL / 4, 128, 0, stream>>>(x, n1g, n1b, xn_h);

    // 3. QKV GEMM: (131072 x 256) @ (256 x 768)
    gemm_wmma_kernel<0, 1, 0><<<dim3(kBL / 128, 768 / 128), 256, 0, stream>>>(
        xn_h, wqkv_t, qkv_b, nullptr, qkv_h, kBL, 768, 256);

    // 4. window attention (2048 windows x 8 heads, 2 window-heads per block)
    attn_kernel<<<8192, 64, 0, stream>>>(qkv_h, relb, attn_h);

    // 5. proj GEMM -> f32
    gemm_wmma_kernel<0, 0, 0><<<dim3(kBL / 128, 256 / 128), 256, 0, stream>>>(
        attn_h, wproj_t, proj_b, nullptr, proj_f, kBL, 256, 256);

    // 6. LN2 -> f16 (reuse xn_h)
    ln_f16_kernel<<<kBL / 4, 128, 0, stream>>>(proj_f, n2g, n2b, xn_h);

    // 7. lin1 GEMM + GELU
    gemm_wmma_kernel<1, 1, 0><<<dim3(kBL / 128, 1024 / 128), 256, 0, stream>>>(
        xn_h, wl1_t, l1b, nullptr, y1_h, kBL, 1024, 256);

    // 8. depthwise 3x3 + GELU
    dwconv_gelu_kernel<<<(size_t)kBL * 1024 / 256, 256, 0, stream>>>(y1_h, dww, dwb, y2_h);

    // 9. lin2 GEMM + residual -> d_out (f32)
    gemm_wmma_kernel<0, 0, 1><<<dim3(kBL / 128, 256 / 128), 256, 0, stream>>>(
        y2_h, wl2_t, l2b, x, out, kBL, 256, 1024);
}
